// EncoderBlock_68788196212938
// MI455X (gfx1250) — matrix operand order repaired and checked by reading, not compiled, not run on hardware
//
#include <hip/hip_runtime.h>
#include <hip/hip_bf16.h>

typedef __attribute__((ext_vector_type(16))) _Float16 v16h;
typedef __attribute__((ext_vector_type(8)))  _Float16 v8h;
typedef __attribute__((ext_vector_type(8)))  float    v8f;
typedef int v4i __attribute__((ext_vector_type(4)));

#if defined(__AMDGCN__) && __has_builtin(__builtin_amdgcn_global_load_async_to_lds_b128)
#define USE_ASYNC 1
typedef __attribute__((address_space(1))) v4i as1_v4i;
typedef __attribute__((address_space(3))) v4i as3_v4i;
__device__ __forceinline__ void async_cp16(const _Float16* g, _Float16* l) {
  __builtin_amdgcn_global_load_async_to_lds_b128((as1_v4i*)g, (as3_v4i*)l, 0, 0);
}
#if __has_builtin(__builtin_amdgcn_s_wait_asynccnt)
#define WAIT_ASYNC(n) __builtin_amdgcn_s_wait_asynccnt(n)
#else
#define WAIT_ASYNC(n) asm volatile("s_wait_asynccnt %0" :: "n"(n) : "memory")
#endif
#else
#define USE_ASYNC 0
#endif

// ---------------------------------------------------------------------------
// WMMA fragment loaders (CDNA5 16x16x32 f16 layouts, wave32)
// A 16x32 (MxK): lane m = lane&15; half0 K={0..7,16..23}, half1 K={8..15,24..31}
// B 32x16 (KxN): lane n = lane&15; half0 K={0..7,16..23}, half1 K={8..15,24..31}
// B staged TRANSPOSED: Bt[n][k] -> each lane reads two 8-half runs of its row, 16 halfs apart.
// ---------------------------------------------------------------------------
__device__ __forceinline__ v16h frag_a(const _Float16* base, int ld, int lane) {
  int m  = lane & 15;
  int k0 = (lane >> 4) << 3;
  const _Float16* p = base + m * ld + k0;
  v8h lo = *(const v8h*)(p);
  v8h hi = *(const v8h*)(p + 16);
  return __builtin_shufflevector(lo, hi, 0,1,2,3,4,5,6,7,8,9,10,11,12,13,14,15);
}
__device__ __forceinline__ v16h frag_b(const _Float16* baseT, int ld, int lane) {
  int n  = lane & 15;
  int k0 = (lane >> 4) << 3;
  const _Float16* p = baseT + n * ld + k0;
  v8h lo = *(const v8h*)(p);
  v8h hi = *(const v8h*)(p + 16);
  return __builtin_shufflevector(lo, hi, 0,1,2,3,4,5,6,7,8,9,10,11,12,13,14,15);
}
__device__ __forceinline__ v8f wmma_f16(v16h a, v16h b, v8f c) {
  return __builtin_amdgcn_wmma_f32_16x16x32_f16(false, a, false, b, (short)0, c, false, false);
}

// ---------------------------------------------------------------------------
// Transposing f32 -> f16 weight conversion: in [R][C] -> out [C][R], batched z.
// block (32,8), grid (C/32, R/32, batches)
// ---------------------------------------------------------------------------
__global__ void cvt_transpose_kernel(const float* __restrict__ in, _Float16* __restrict__ out,
                                     int R, int C) {
  __shared__ float tile[32][33];
  long batch = blockIdx.z;
  const float* ip = in + batch * (long)R * C;
  _Float16* op = out + batch * (long)R * C;
  int c0 = blockIdx.x * 32, r0 = blockIdx.y * 32;
  int tc = threadIdx.x, tr = threadIdx.y;
#pragma unroll
  for (int i = 0; i < 4; ++i)
    tile[tr + i * 8][tc] = ip[(long)(r0 + tr + i * 8) * C + c0 + tc];
  __syncthreads();
#pragma unroll
  for (int i = 0; i < 4; ++i)
    op[(long)(c0 + tr + i * 8) * R + r0 + tc] = (_Float16)tile[tc][tr + i * 8];
}

// ---------------------------------------------------------------------------
// LayerNorm: one row (E=1024) per 256-thread block, f16 output
// ---------------------------------------------------------------------------
__global__ void ln_f16_kernel(const float* __restrict__ x, const float* __restrict__ w,
                              const float* __restrict__ bb, _Float16* __restrict__ out, int E) {
  int row = blockIdx.x;
  const float* xr = x + (size_t)row * E;
  int t = threadIdx.x, lane = t & 31, wid = t >> 5;
  float v[4]; float s = 0.f;
#pragma unroll
  for (int i = 0; i < 4; ++i) { v[i] = xr[t + i * 256]; s += v[i]; }
#pragma unroll
  for (int off = 16; off; off >>= 1) s += __shfl_xor(s, off, 32);
  __shared__ float red[8];
  if (lane == 0) red[wid] = s;
  __syncthreads();
  float tot = 0.f;
#pragma unroll
  for (int i = 0; i < 8; ++i) tot += red[i];
  float mu = tot / (float)E;
  float vs = 0.f;
#pragma unroll
  for (int i = 0; i < 4; ++i) { float dd = v[i] - mu; vs += dd * dd; }
#pragma unroll
  for (int off = 16; off; off >>= 1) vs += __shfl_xor(vs, off, 32);
  __syncthreads();
  if (lane == 0) red[wid] = vs;
  __syncthreads();
  float vt = 0.f;
#pragma unroll
  for (int i = 0; i < 8; ++i) vt += red[i];
  float rstd = rsqrtf(vt / (float)E + 1e-5f);
#pragma unroll
  for (int i = 0; i < 4; ++i) {
    int c = t + i * 256;
    out[(size_t)row * E + c] = (_Float16)((v[i] - mu) * rstd * w[c] + bb[c]);
  }
}

#define BK 32
#define LDA 40
#define LDB 40

// ---------------------------------------------------------------------------
// Big WMMA GEMM: C[128x128 tile] = A[M,K] x B[K,N], B given TRANSPOSED Bt[N][K].
// 8 waves; wave = 32-row strip x 64-col half => 2x4 sub-tiles, 8 WMMA / K-step.
// Double-buffered LDS, async-to-LDS (4 b128 chunks/thread/tile).
// EPI 0: f32 = acc + res | EPI 1: f16 = relu(acc+bias) | EPI 2: f32 = acc+bias+res
// ---------------------------------------------------------------------------
template<int EPI>
__global__ void gemm_wmma(const _Float16* __restrict__ A, const _Float16* __restrict__ Bt,
                          int N, int K,
                          const float* __restrict__ bias, const float* __restrict__ res,
                          float* __restrict__ of32, _Float16* __restrict__ of16) {
  __shared__ alignas(16) _Float16 As[2][128 * LDA];
  __shared__ alignas(16) _Float16 Bs[2][128 * LDB];
  int t = threadIdx.x, lane = t & 31, ws = t >> 5;
  int mw = ws & 3, nw = ws >> 2;              // wave: rows mw*32..+32, cols nw*64..+64
  long bm = (long)blockIdx.x * 128, bn = (long)blockIdx.y * 128;
  int r4 = t >> 2, c8 = (t & 3) << 3;
  const _Float16* gA0 = &A[(bm + r4) * (long)K + c8];
  const _Float16* gA1 = &A[(bm + 64 + r4) * (long)K + c8];
  const _Float16* gB0 = &Bt[(bn + r4) * (long)K + c8];
  const _Float16* gB1 = &Bt[(bn + 64 + r4) * (long)K + c8];
  v8f acc[2][4] = {};
  int nkt = K / BK;

#if USE_ASYNC
  async_cp16(gA0, &As[0][r4 * LDA + c8]);
  async_cp16(gA1, &As[0][(64 + r4) * LDA + c8]);
  async_cp16(gB0, &Bs[0][r4 * LDB + c8]);
  async_cp16(gB1, &Bs[0][(64 + r4) * LDB + c8]);
  for (int i = 0; i < nkt; ++i) {
    int cur = i & 1;
    if (i + 1 < nkt) {
      int kt = (i + 1) * BK;
      async_cp16(gA0 + kt, &As[cur ^ 1][r4 * LDA + c8]);
      async_cp16(gA1 + kt, &As[cur ^ 1][(64 + r4) * LDA + c8]);
      async_cp16(gB0 + kt, &Bs[cur ^ 1][r4 * LDB + c8]);
      async_cp16(gB1 + kt, &Bs[cur ^ 1][(64 + r4) * LDB + c8]);
      WAIT_ASYNC(4);
    } else {
      WAIT_ASYNC(0);
    }
    __syncthreads();
    v16h a0 = frag_a(&As[cur][(mw * 32) * LDA], LDA, lane);
    v16h a1 = frag_a(&As[cur][(mw * 32 + 16) * LDA], LDA, lane);
#pragma unroll
    for (int nt = 0; nt < 4; ++nt) {
      v16h b = frag_b(&Bs[cur][(nw * 64 + nt * 16) * LDB], LDB, lane);
      acc[0][nt] = wmma_f16(a0, b, acc[0][nt]);
      acc[1][nt] = wmma_f16(a1, b, acc[1][nt]);
    }
    __syncthreads();
  }
#else
  v8h ra0 = *(const v8h*)gA0, ra1 = *(const v8h*)gA1;
  v8h rb0 = *(const v8h*)gB0, rb1 = *(const v8h*)gB1;
  for (int i = 0; i < nkt; ++i) {
    int cur = i & 1;
    *(v8h*)&As[cur][r4 * LDA + c8] = ra0;
    *(v8h*)&As[cur][(64 + r4) * LDA + c8] = ra1;
    *(v8h*)&Bs[cur][r4 * LDB + c8] = rb0;
    *(v8h*)&Bs[cur][(64 + r4) * LDB + c8] = rb1;
    if (i + 1 < nkt) {
      int kt = (i + 1) * BK;
      ra0 = *(const v8h*)(gA0 + kt);
      ra1 = *(const v8h*)(gA1 + kt);
      rb0 = *(const v8h*)(gB0 + kt);
      rb1 = *(const v8h*)(gB1 + kt);
    }
    __syncthreads();
    v16h a0 = frag_a(&As[cur][(mw * 32) * LDA], LDA, lane);
    v16h a1 = frag_a(&As[cur][(mw * 32 + 16) * LDA], LDA, lane);
#pragma unroll
    for (int nt = 0; nt < 4; ++nt) {
      v16h b = frag_b(&Bs[cur][(nw * 64 + nt * 16) * LDB], LDB, lane);
      acc[0][nt] = wmma_f16(a0, b, acc[0][nt]);
      acc[1][nt] = wmma_f16(a1, b, acc[1][nt]);
    }
    __syncthreads();
  }
#endif

  int hf = lane >> 4, n0 = lane & 15;
#pragma unroll
  for (int mt = 0; mt < 2; ++mt)
#pragma unroll
    for (int nt = 0; nt < 4; ++nt)
#pragma unroll
      for (int r = 0; r < 8; ++r) {
        long row = bm + mw * 32 + mt * 16 + r + hf * 8;
        long col = bn + nw * 64 + nt * 16 + n0;
        long idx = row * N + col;
        float v = acc[mt][nt][r];
        if (EPI == 0)      of32[idx] = v + res[idx];
        else if (EPI == 1) { v += bias[col]; of16[idx] = (_Float16)(v > 0.f ? v : 0.f); }
        else               of32[idx] = v + bias[col] + res[idx];
      }
}

// ---------------------------------------------------------------------------
// Per-head Q/K/V projection (N=64 tile). Weights pre-transposed: Wt[h]=[64][E].
// grid=(Ntok/128, H, 3). Q/K -> [B,H,S,64] (Q scaled 1/8); V -> [B,H,64,S].
// ---------------------------------------------------------------------------
__global__ void qkv_gemm(const _Float16* __restrict__ h,
                         const _Float16* __restrict__ Wqt, const _Float16* __restrict__ Wkt,
                         const _Float16* __restrict__ Wvt,
                         _Float16* __restrict__ Qo, _Float16* __restrict__ Ko, _Float16* __restrict__ Vt,
                         int S, int E, int H) {
  const int d = 64;
  __shared__ alignas(16) _Float16 As[2][128 * LDA];
  __shared__ alignas(16) _Float16 Bs[2][64 * LDB];
  int t = threadIdx.x, lane = t & 31, ws = t >> 5;
  int head = blockIdx.y, sel = blockIdx.z;
  const _Float16* W = (sel == 0 ? Wqt : sel == 1 ? Wkt : Wvt) + (long)head * E * d;
  float scale = (sel == 0) ? 0.125f : 1.0f;
  long bm = (long)blockIdx.x * 128;
  int r4 = t >> 2, c8 = (t & 3) << 3;
  const _Float16* gA0 = &h[(bm + r4) * (long)E + c8];
  const _Float16* gA1 = &h[(bm + 64 + r4) * (long)E + c8];
  const _Float16* gB  = &W[(long)r4 * E + c8];
  v8f acc[4] = {};
  int nkt = E / BK;

#if USE_ASYNC
  async_cp16(gA0, &As[0][r4 * LDA + c8]);
  async_cp16(gA1, &As[0][(64 + r4) * LDA + c8]);
  async_cp16(gB,  &Bs[0][r4 * LDB + c8]);
  for (int i = 0; i < nkt; ++i) {
    int cur = i & 1;
    if (i + 1 < nkt) {
      int kt = (i + 1) * BK;
      async_cp16(gA0 + kt, &As[cur ^ 1][r4 * LDA + c8]);
      async_cp16(gA1 + kt, &As[cur ^ 1][(64 + r4) * LDA + c8]);
      async_cp16(gB  + kt, &Bs[cur ^ 1][r4 * LDB + c8]);
      WAIT_ASYNC(3);
    } else {
      WAIT_ASYNC(0);
    }
    __syncthreads();
    v16h a = frag_a(&As[cur][ws * 16 * LDA], LDA, lane);
#pragma unroll
    for (int nt = 0; nt < 4; ++nt) {
      v16h b = frag_b(&Bs[cur][nt * 16 * LDB], LDB, lane);
      acc[nt] = wmma_f16(a, b, acc[nt]);
    }
    __syncthreads();
  }
#else
  v8h ra0 = *(const v8h*)gA0, ra1 = *(const v8h*)gA1, rb = *(const v8h*)gB;
  for (int i = 0; i < nkt; ++i) {
    int cur = i & 1;
    *(v8h*)&As[cur][r4 * LDA + c8] = ra0;
    *(v8h*)&As[cur][(64 + r4) * LDA + c8] = ra1;
    *(v8h*)&Bs[cur][r4 * LDB + c8] = rb;
    if (i + 1 < nkt) {
      int kt = (i + 1) * BK;
      ra0 = *(const v8h*)(gA0 + kt);
      ra1 = *(const v8h*)(gA1 + kt);
      rb  = *(const v8h*)(gB + kt);
    }
    __syncthreads();
    v16h a = frag_a(&As[cur][ws * 16 * LDA], LDA, lane);
#pragma unroll
    for (int nt = 0; nt < 4; ++nt) {
      v16h b = frag_b(&Bs[cur][nt * 16 * LDB], LDB, lane);
      acc[nt] = wmma_f16(a, b, acc[nt]);
    }
    __syncthreads();
  }
#endif

  int hf = lane >> 4, n0 = lane & 15;
#pragma unroll
  for (int nt = 0; nt < 4; ++nt)
#pragma unroll
    for (int r = 0; r < 8; ++r) {
      long tok = bm + ws * 16 + r + hf * 8;
      long b = tok / S, s = tok % S;
      float v = acc[nt][r] * scale;
      if (sel < 2) {
        _Float16* O = (sel == 0) ? Qo : Ko;
        O[((b * H + head) * S + s) * d + nt * 16 + n0] = (_Float16)v;
      } else {
        Vt[((b * H + head) * d + nt * 16 + n0) * S + s] = (_Float16)v;
      }
    }
}

// ---------------------------------------------------------------------------
// Flash attention: grid=(S/128, B*H). Q pre-scaled; V transposed [B,H,64,S].
// Output concat layout [B,S,E] f16. Double-buffered K/V tiles (2 chunks/thread).
// ---------------------------------------------------------------------------
__global__ void attn_kernel(const _Float16* __restrict__ Qf, const _Float16* __restrict__ Kf,
                            const _Float16* __restrict__ Vt, _Float16* __restrict__ O,
                            int S, int H) {
  const int d = 64, E = 64 * 16;
  __shared__ alignas(16) _Float16 Qs[128 * 72];
  __shared__ alignas(16) _Float16 Ks[2][32 * 72];
  __shared__ alignas(16) _Float16 Vs[2][64 * 40];
  __shared__ alignas(16) _Float16 Ps[8 * 16 * 40];
  int t = threadIdx.x, lane = t & 31, ws = t >> 5;
  int bh = blockIdx.y;
  int b = bh / H, hh = bh % H;
  long kbase = (long)bh * S;          // K rows base [B*H*S]
  long vbase = (long)bh * d;          // Vt rows base [B*H*d]
  int q0 = blockIdx.x * 128;
  int kr = t >> 3, kc = (t & 7) << 3; // K tile: 32 rows x 64
  int vr = t >> 2, vc = (t & 3) << 3; // V tile: 64 rows x 32
  const _Float16* gK = &Kf[(kbase + kr) * d + kc];
  const _Float16* gV = &Vt[(vbase + vr) * (long)S + vc];

  float m_r[8], l_r[8];
  v8f acc_o[4] = {};
#pragma unroll
  for (int r = 0; r < 8; ++r) { m_r[r] = -1e30f; l_r[r] = 0.f; }
  int njt = S / 32;

#if USE_ASYNC
#pragma unroll
  for (int i = 0; i < 4; ++i) {
    int e = t + i * 256;
    int row = e >> 3, col = (e & 7) << 3;
    async_cp16(&Qf[(kbase + q0 + row) * d + col], &Qs[row * 72 + col]);
  }
  async_cp16(gK, &Ks[0][kr * 72 + kc]);
  async_cp16(gV, &Vs[0][vr * 40 + vc]);
#else
#pragma unroll
  for (int i = 0; i < 4; ++i) {
    int e = t + i * 256;
    int row = e >> 3, col = (e & 7) << 3;
    *(v8h*)&Qs[row * 72 + col] = *(const v8h*)&Qf[(kbase + q0 + row) * d + col];
  }
  v8h rk = *(const v8h*)gK, rv = *(const v8h*)gV;
#endif

  for (int j = 0; j < njt; ++j) {
    int cur = j & 1;
#if USE_ASYNC
    if (j + 1 < njt) {
      async_cp16(gK + (long)(j + 1) * 32 * d, &Ks[cur ^ 1][kr * 72 + kc]);
      async_cp16(gV + (j + 1) * 32,           &Vs[cur ^ 1][vr * 40 + vc]);
      WAIT_ASYNC(2);
    } else {
      WAIT_ASYNC(0);
    }
#else
    *(v8h*)&Ks[cur][kr * 72 + kc] = rk;
    *(v8h*)&Vs[cur][vr * 40 + vc] = rv;
    if (j + 1 < njt) {
      rk = *(const v8h*)(gK + (long)(j + 1) * 32 * d);
      rv = *(const v8h*)(gV + (j + 1) * 32);
    }
#endif
    __syncthreads();
    // scores 16x32 per wave
    v8f sc[2] = {};
#pragma unroll
    for (int kp = 0; kp < 2; ++kp) {
      v16h a = frag_a(&Qs[ws * 16 * 72 + kp * 32], 72, lane);
#pragma unroll
      for (int nt = 0; nt < 2; ++nt) {
        v16h bb = frag_b(&Ks[cur][nt * 16 * 72 + kp * 32], 72, lane);
        sc[nt] = wmma_f16(a, bb, sc[nt]);
      }
    }
    // online softmax (rows live in one 16-lane half-wave; elem r <-> row r+8*half)
    float mx[8];
#pragma unroll
    for (int r = 0; r < 8; ++r) mx[r] = fmaxf(sc[0][r], sc[1][r]);
#pragma unroll
    for (int off = 1; off < 16; off <<= 1)
#pragma unroll
      for (int r = 0; r < 8; ++r) mx[r] = fmaxf(mx[r], __shfl_xor(mx[r], off, 32));
    float scl[8], sum[8];
#pragma unroll
    for (int r = 0; r < 8; ++r) {
      float mn = fmaxf(m_r[r], mx[r]);
      scl[r] = __expf(m_r[r] - mn);
      m_r[r] = mn;
    }
#pragma unroll
    for (int r = 0; r < 8; ++r) {
      sc[0][r] = __expf(sc[0][r] - m_r[r]);
      sc[1][r] = __expf(sc[1][r] - m_r[r]);
      sum[r] = sc[0][r] + sc[1][r];
    }
#pragma unroll
    for (int off = 1; off < 16; off <<= 1)
#pragma unroll
      for (int r = 0; r < 8; ++r) sum[r] += __shfl_xor(sum[r], off, 32);
#pragma unroll
    for (int r = 0; r < 8; ++r) l_r[r] = l_r[r] * scl[r] + sum[r];
#pragma unroll
    for (int nt = 0; nt < 4; ++nt)
#pragma unroll
      for (int r = 0; r < 8; ++r) acc_o[nt][r] *= scl[r];
    // P: C-layout -> A-layout via per-wave LDS round trip
    int hf = lane >> 4, n0 = lane & 15;
    _Float16* Pw = &Ps[ws * 16 * 40];
#pragma unroll
    for (int nt = 0; nt < 2; ++nt)
#pragma unroll
      for (int r = 0; r < 8; ++r)
        Pw[(r + hf * 8) * 40 + nt * 16 + n0] = (_Float16)sc[nt][r];
    v16h pa = frag_a(Pw, 40, lane);
#pragma unroll
    for (int nt = 0; nt < 4; ++nt) {
      v16h bb = frag_b(&Vs[cur][nt * 16 * 40], 40, lane);
      acc_o[nt] = wmma_f16(pa, bb, acc_o[nt]);
    }
    __syncthreads();
  }
  int hf = lane >> 4, n0 = lane & 15;
#pragma unroll
  for (int nt = 0; nt < 4; ++nt)
#pragma unroll
    for (int r = 0; r < 8; ++r) {
      long s = q0 + ws * 16 + r + hf * 8;
      long col = (long)hh * 64 + nt * 16 + n0;
      O[((long)b * S + s) * E + col] = (_Float16)(acc_o[nt][r] / l_r[r]);
    }
}

// ---------------------------------------------------------------------------
// Host side
// ---------------------------------------------------------------------------
extern "C" void kernel_launch(void* const* d_in, const int* in_sizes, int n_in,
                              void* d_out, int out_size, void* d_ws, size_t ws_size,
                              hipStream_t stream) {
  const int B = 2, S = 2048, E = 1024, H = 16, F = 4096, d = 64;
  const long Ntok = (long)B * S;

  const float* x     = (const float*)d_in[0];
  const float* Wq    = (const float*)d_in[1];
  const float* Wk    = (const float*)d_in[2];
  const float* Wv    = (const float*)d_in[3];
  const float* Wo    = (const float*)d_in[4];
  const float* ln1_w = (const float*)d_in[5];
  const float* ln1_b = (const float*)d_in[6];
  const float* ln2_w = (const float*)d_in[7];
  const float* ln2_b = (const float*)d_in[8];
  const float* W1    = (const float*)d_in[9];
  const float* b1    = (const float*)d_in[10];
  const float* W2    = (const float*)d_in[11];
  const float* b2    = (const float*)d_in[12];
  float* out = (float*)d_out;

  char* wp = (char*)d_ws;
  auto carve = [&](size_t bytes) { char* p = wp; wp += (bytes + 255) & ~(size_t)255; return p; };
  _Float16* h16   = (_Float16*)carve(Ntok * E * 2);
  _Float16* Wqt   = (_Float16*)carve((size_t)H * E * d * 2);
  _Float16* Wkt   = (_Float16*)carve((size_t)H * E * d * 2);
  _Float16* Wvt   = (_Float16*)carve((size_t)H * E * d * 2);
  _Float16* Wot   = (_Float16*)carve((size_t)E * E * 2);
  _Float16* W1t   = (_Float16*)carve((size_t)E * F * 2);
  _Float16* W2t   = (_Float16*)carve((size_t)F * E * 2);
  _Float16* Qf    = (_Float16*)carve((size_t)B * H * S * d * 2);
  _Float16* Kf    = (_Float16*)carve((size_t)B * H * S * d * 2);
  _Float16* Vtb   = (_Float16*)carve((size_t)B * H * S * d * 2);
  _Float16* attn16= (_Float16*)carve(Ntok * E * 2);
  float*    x1    = (float*)   carve(Ntok * E * 4);
  _Float16* h216  = (_Float16*)carve(Ntok * E * 2);
  _Float16* ff16  = (_Float16*)carve(Ntok * F * 2);

  dim3 blk(256);
  dim3 tblk(32, 8);

  // transposing weight conversions f32 -> f16 (Bt layouts)
  cvt_transpose_kernel<<<dim3(d / 32, E / 32, H), tblk, 0, stream>>>(Wq, Wqt, E, d);
  cvt_transpose_kernel<<<dim3(d / 32, E / 32, H), tblk, 0, stream>>>(Wk, Wkt, E, d);
  cvt_transpose_kernel<<<dim3(d / 32, E / 32, H), tblk, 0, stream>>>(Wv, Wvt, E, d);
  cvt_transpose_kernel<<<dim3(E / 32, E / 32, 1), tblk, 0, stream>>>(Wo, Wot, E, E);
  cvt_transpose_kernel<<<dim3(F / 32, E / 32, 1), tblk, 0, stream>>>(W1, W1t, E, F);
  cvt_transpose_kernel<<<dim3(E / 32, F / 32, 1), tblk, 0, stream>>>(W2, W2t, F, E);

  // LN1 -> h16
  ln_f16_kernel<<<dim3((unsigned)Ntok), blk, 0, stream>>>(x, ln1_w, ln1_b, h16, E);

  // QKV projections (Q pre-scaled by 1/sqrt(d); V stored transposed)
  qkv_gemm<<<dim3(Ntok / 128, H, 3), blk, 0, stream>>>(h16, Wqt, Wkt, Wvt, Qf, Kf, Vtb, S, E, H);

  // attention -> attn16 [B,S,E]
  attn_kernel<<<dim3(S / 128, B * H), blk, 0, stream>>>(Qf, Kf, Vtb, attn16, S, H);

  // x1 = x + attn @ Wo   (128x128 tiles)
  gemm_wmma<0><<<dim3(Ntok / 128, E / 128), blk, 0, stream>>>(attn16, Wot, E, E, nullptr, x, x1, nullptr);

  // LN2 -> h216
  ln_f16_kernel<<<dim3((unsigned)Ntok), blk, 0, stream>>>(x1, ln2_w, ln2_b, h216, E);

  // ff = relu(h2 @ W1 + b1)
  gemm_wmma<1><<<dim3(Ntok / 128, F / 128), blk, 0, stream>>>(h216, W1t, F, E, b1, nullptr, nullptr, ff16);

  // out = x1 + ff @ W2 + b2
  gemm_wmma<2><<<dim3(Ntok / 128, E / 128), blk, 0, stream>>>(ff16, W2t, E, F, b2, x1, out, nullptr);
}
